// MGN_NET_39779987096422
// MI455X (gfx1250) — compile-verified, matrix operand order checked by reading, not run on hardware
//
#include <hip/hip_runtime.h>
#include <hip/hip_bf16.h>

#define N_NODES 35
#define N_EDGES 1190
#define N_VIEWS 6

typedef __attribute__((ext_vector_type(16))) __bf16 v16bf;
typedef __attribute__((ext_vector_type(8)))  __bf16 v8bf;
typedef __attribute__((ext_vector_type(8)))  float  v8f;

// ---------------------------------------------------------------------------
// Prep: pack one MLP row (6 f32 weights + f32 bias) into 16 bytes of bf16:
//   out[r] = { w0..w5, bias, 0 }   (bias==nullptr -> 1.0, used for edge_attr)
// This turns the per-iteration B-fragment build in the hot loop into a single
// 16B global_load_b128 with no conversions, and lets the WMMA itself apply the
// bias (A carries 1.0 at K=6).
// ---------------------------------------------------------------------------
__global__ void pack_rows_bf16(const float* __restrict__ w,     // [R, 6]
                               const float* __restrict__ bias,  // [R] or null
                               v8bf* __restrict__ out, int R)
{
    const int r = blockIdx.x * 256 + threadIdx.x;
    if (r >= R) return;
    v8bf h;
    #pragma unroll
    for (int k = 0; k < N_VIEWS; ++k) h[k] = (__bf16)w[r * N_VIEWS + k];
    h[6] = (__bf16)(bias ? bias[r] : 1.0f);
    h[7] = (__bf16)0.f;
    out[r] = h;
}

// ---------------------------------------------------------------------------
// Fused NNConv message kernel (V_WMMA_F32_16X16X32_BF16).
//
//   msg[e, o] = sum_i x[src[e], i] * relu( ea[e,:] . nnw[i*C_OUT+o, :] + nnb[..] )
//
// One wave owns a [16 edges x 16 out-channels] tile. Per input channel i, one
// bf16 WMMA (K = 7 used of 32: 6 views + bias lane) produces the biased
// pre-activation tile; ReLU + per-row scale by x[src, i] accumulate in
// registers. The [E, C_IN, C_OUT] per-edge weight tensor (312 MB for layer 2
// in the reference) is never materialized.
//
// Fragment packing (ISA 7.12.2, wave32):
//   A 16x32 bf16 : lanes 0-15, elems 0..7 = K0..7 -> packed {ea, 1, 0}
//   B 32x16 bf16 : lanes 0-15, elems 0..7 = K0..7 -> packed {w, bias, 0}
//   upper-half lanes / remaining K are zero quads (pad contributes nothing)
//   C 16x16 f32  : row = vgpr_j + 8*(lane>=16), col = lane&15
// ---------------------------------------------------------------------------
template<int C_IN, int C_OUT>
__global__ __launch_bounds__(128)
void nnconv_msg_wmma(const float* __restrict__ x_prev,   // [N_NODES, C_IN]
                     const v8bf*  __restrict__ eap,      // [N_EDGES] packed
                     const int*   __restrict__ src,      // [N_EDGES]
                     const v8bf*  __restrict__ nnwp,     // [C_IN*C_OUT] packed
                     float* __restrict__ msg)            // [N_EDGES, C_OUT]
{
    // xs[i*16 + row] = x_prev[src[eBase+row], i]  (column-major: fixed-i reads
    // touch 16 consecutive LDS words -> conflict-free broadcast across lanes)
    __shared__ float xs[C_IN * 16];

    const int eBase = blockIdx.x * 16;
    const int tid   = threadIdx.x;

    for (int idx = tid; idx < C_IN * 16; idx += 128) {
        const int i = idx >> 4, row = idx & 15;
        const int e = eBase + row;
        float v = 0.f;
        if (e < N_EDGES) v = x_prev[src[e] * C_IN + i];
        xs[idx] = v;
    }
    __syncthreads();

    const int lane   = tid & 31;
    const int wave   = tid >> 5;
    const int laneLo = lane & 15;
    const int laneHi = lane >> 4;
    const int oBase  = (blockIdx.y * 4 + wave) * 16;

    union Frag { v16bf v; uint4 q[2]; };
    const uint4 z4 = {0u, 0u, 0u, 0u};

    // A fragment: constant across the whole i-loop.
    Frag fa;
    fa.q[0] = z4; fa.q[1] = z4;
    if (laneHi == 0) {
        const int e = eBase + laneLo;
        if (e < N_EDGES) fa.q[0] = ((const uint4*)eap)[e];
    }

    float acc[8];
    #pragma unroll
    for (int j = 0; j < 8; ++j) acc[j] = 0.f;

    const uint4* nq = (const uint4*)nnwp;
    int rIdx = oBase + laneLo;                 // packed-row index for i = 0

    #pragma unroll 4
    for (int i = 0; i < C_IN; ++i, rIdx += C_OUT) {
        uint4 t = z4;
        if (laneHi == 0) {
            t = nq[rIdx];                      // one global_load_b128
            if (i + 1 < C_IN)                  // pull next i's row toward WGP
                __builtin_prefetch((const void*)(nq + rIdx + C_OUT), 0, 0);
        }
        Frag fb;
        fb.q[0] = t; fb.q[1] = z4;

        v8f c;
        #pragma unroll
        for (int j = 0; j < 8; ++j) c[j] = 0.f;
        c = __builtin_amdgcn_wmma_f32_16x16x32_bf16(
                /*neg_a=*/false, fa.v, /*neg_b=*/false, fb.v,
                /*c_mod=*/(short)0, c, /*reuse_a=*/false, /*reuse_b=*/false);

        #pragma unroll
        for (int j = 0; j < 8; ++j) {
            const float w = fmaxf(c[j], 0.f);          // relu(dot + bias)
            acc[j] = fmaf(xs[i * 16 + j + 8 * laneHi], w, acc[j]);
        }
    }

    #pragma unroll
    for (int j = 0; j < 8; ++j) {
        const int e = eBase + j + 8 * laneHi;
        if (e < N_EDGES) msg[e * C_OUT + oBase + laneLo] = acc[j];
    }
}

// ---------------------------------------------------------------------------
// Deterministic scatter-mean + root transform + bias + ReLU.
// One block per destination node (no float atomics -> bitwise deterministic).
// blockDim.x == C_OUT.
// ---------------------------------------------------------------------------
template<int C_IN, int C_OUT>
__global__ void nnconv_aggregate(const float* __restrict__ msg,     // [E, C_OUT]
                                 const int*   __restrict__ dst,     // [E]
                                 const float* __restrict__ x_prev,  // [N, C_IN]
                                 const float* __restrict__ lin_w,   // [C_OUT, C_IN]
                                 const float* __restrict__ bias,    // [C_OUT]
                                 float* __restrict__ x_out)         // [N, C_OUT]
{
    __shared__ float xp[C_IN];
    const int n = blockIdx.x;
    const int o = threadIdx.x;

    for (int k = o; k < C_IN; k += C_OUT) xp[k] = x_prev[n * C_IN + k];
    __syncthreads();

    float s = 0.f, cnt = 0.f;
    for (int e = 0; e < N_EDGES; ++e) {
        if (dst[e] == n) { s += msg[e * C_OUT + o]; cnt += 1.f; }
    }
    s /= fmaxf(cnt, 1.f);

    float rt = 0.f;
    for (int k = 0; k < C_IN; ++k)
        rt = fmaf(xp[k], lin_w[o * C_IN + k], rt);

    x_out[n * C_OUT + o] = fmaxf(s + rt + bias[o], 0.f);
}

// ---------------------------------------------------------------------------
// CBT: out[a,b] = sum_c |x[a,c] - x[b,c]|   (x = [35, 64] post-ReLU features)
// ---------------------------------------------------------------------------
__global__ void cbt_kernel(const float* __restrict__ x, float* __restrict__ out)
{
    const int id = blockIdx.x * blockDim.x + threadIdx.x;
    if (id >= N_NODES * N_NODES) return;
    const int a = id / N_NODES, b = id % N_NODES;
    float s = 0.f;
    #pragma unroll 8
    for (int c = 0; c < 64; ++c)
        s += fabsf(x[a * 64 + c] - x[b * 64 + c]);
    out[id] = s;
}

// ---------------------------------------------------------------------------
extern "C" void kernel_launch(void* const* d_in, const int* in_sizes, int n_in,
                              void* d_out, int out_size, void* d_ws, size_t ws_size,
                              hipStream_t stream)
{
    (void)in_sizes; (void)n_in; (void)out_size; (void)ws_size;

    const float* x0   = (const float*)d_in[0];   // [35, 1]
    const float* ea   = (const float*)d_in[1];   // [1190, 6]
    const int*   eidx = (const int*)  d_in[2];   // [2, 1190]
    const float* nn1w = (const float*)d_in[3];
    const float* nn1b = (const float*)d_in[4];
    const float* lin1 = (const float*)d_in[5];
    const float* b1   = (const float*)d_in[6];
    const float* nn2w = (const float*)d_in[7];
    const float* nn2b = (const float*)d_in[8];
    const float* lin2 = (const float*)d_in[9];
    const float* b2   = (const float*)d_in[10];
    const float* nn3w = (const float*)d_in[11];
    const float* nn3b = (const float*)d_in[12];
    const float* lin3 = (const float*)d_in[13];
    const float* b3   = (const float*)d_in[14];

    const int* src = eidx;            // edge_index[0]
    const int* dst = eidx + N_EDGES;  // edge_index[1]

    // workspace layout (~2.7 MB total, L2-resident)
    char* ws = (char*)d_ws;
    size_t off = 0;
    auto alloc = [&](size_t bytes) {
        void* p = ws + off;
        off = (off + bytes + 255) & ~(size_t)255;
        return p;
    };
    float* msg  = (float*)alloc((size_t)N_EDGES * 256 * sizeof(float));
    v8bf*  nn1p = (v8bf*) alloc((size_t)1 * 256 * 16);
    v8bf*  nn2p = (v8bf*) alloc((size_t)256 * 256 * 16);
    v8bf*  nn3p = (v8bf*) alloc((size_t)256 * 64 * 16);
    v8bf*  eap  = (v8bf*) alloc((size_t)N_EDGES * 16);
    float* x1   = (float*)alloc((size_t)N_NODES * 256 * sizeof(float));
    float* x2   = (float*)alloc((size_t)N_NODES * 256 * sizeof(float));
    float* x3   = (float*)alloc((size_t)N_NODES * 64 * sizeof(float));

    // pack launch-invariant operands to bf16 (rerun every call: stateless)
    pack_rows_bf16<<<(256 + 255) / 256, 256, 0, stream>>>(nn1w, nn1b, nn1p, 256);
    pack_rows_bf16<<<(65536 + 255) / 256, 256, 0, stream>>>(nn2w, nn2b, nn2p, 65536);
    pack_rows_bf16<<<(16384 + 255) / 256, 256, 0, stream>>>(nn3w, nn3b, nn3p, 16384);
    pack_rows_bf16<<<(N_EDGES + 255) / 256, 256, 0, stream>>>(ea, nullptr, eap, N_EDGES);

    const int eTiles = (N_EDGES + 15) / 16;  // 75

    // layer 1 (C_IN=1, C_OUT=256)
    nnconv_msg_wmma<1, 256><<<dim3(eTiles, 4), 128, 0, stream>>>(
        x0, eap, src, nn1p, msg);
    nnconv_aggregate<1, 256><<<N_NODES, 256, 0, stream>>>(
        msg, dst, x0, lin1, b1, x1);

    // layer 2 (C_IN=256, C_OUT=256) — the heavy one
    nnconv_msg_wmma<256, 256><<<dim3(eTiles, 4), 128, 0, stream>>>(
        x1, eap, src, nn2p, msg);
    nnconv_aggregate<256, 256><<<N_NODES, 256, 0, stream>>>(
        msg, dst, x1, lin2, b2, x2);

    // layer 3 (C_IN=256, C_OUT=64)
    nnconv_msg_wmma<256, 64><<<dim3(eTiles, 1), 128, 0, stream>>>(
        x2, eap, src, nn3p, msg);
    nnconv_aggregate<256, 64><<<N_NODES, 64, 0, stream>>>(
        msg, dst, x2, lin3, b3, x3);

    // pairwise L1 output [35, 35]
    cbt_kernel<<<(N_NODES * N_NODES + 255) / 256, 256, 0, stream>>>(
        x3, (float*)d_out);
}
